// MultiheadAdditiveAttention_6356551598438
// MI455X (gfx1250) — compile-verified, matrix operand order checked by reading, not run on hardware
//
#include <hip/hip_runtime.h>

// ---------------------------------------------------------------------------
// MultiheadAdditiveAttention for MI455X (gfx1250, wave32, WMMA)
// B=2, T=S=384, E=768, H=12, HD=64
// ---------------------------------------------------------------------------

typedef __attribute__((ext_vector_type(16))) _Float16 v16h;
typedef __attribute__((ext_vector_type(8)))  _Float16 h8v;
typedef __attribute__((ext_vector_type(8)))  float    v8f;

constexpr int Bc = 2, Tc = 384, Ec = 768, Hc = 12, HDc = 64;
constexpr int Mc = Bc * Tc;   // 768 rows of the [B*T, E] activation matrix
constexpr int Sc = Tc;        // source length
constexpr int SPAD = Sc + 8;  // +8 halves: bank-conflict-free ds_read_b128

// Hardware tanh (V_TANH_F32 on gfx1250) when the builtin exists.
static __device__ inline float fast_tanh(float x) {
#if __has_builtin(__builtin_amdgcn_tanhf)
    return __builtin_amdgcn_tanhf(x);
#elif __has_builtin(__builtin_amdgcn_tanh_f32)
    return __builtin_amdgcn_tanh_f32(x);
#else
    return tanhf(x);
#endif
}

// ----- build a 16x32 (MxK) f16 A/B WMMA fragment from K-contiguous memory ---
// Per CDNA5 ISA 7.12.2 (16-bit A 16x32): lane m = lane&15, khalf = lane>>4.
// v16h elements 0..7  hold k = khalf*8 + {0..7}
// v16h elements 8..15 hold k = khalf*8 + 16 + {0..7}
// => two 16-byte loads at (base) and (base + 16 halves).
static __device__ inline v16h make_frag(const _Float16* p) {
    h8v lo = *reinterpret_cast<const h8v*>(p);
    h8v hi = *reinterpret_cast<const h8v*>(p + 16);
    v16h r;
#pragma unroll
    for (int i = 0; i < 8; ++i) { r[i] = lo[i]; r[i + 8] = hi[i]; }
    return r;
}

// ---------------------------------------------------------------------------
// Kernel 1: f32 -> f16 elementwise convert
// ---------------------------------------------------------------------------
__global__ void k_convert_f16(const float* __restrict__ in,
                              _Float16* __restrict__ out, int n) {
    int i = blockIdx.x * 256 + threadIdx.x;
    if (i < n) out[i] = (_Float16)in[i];
}

// ---------------------------------------------------------------------------
// Kernel 2: weight transpose + convert:  W[k][n] (f32, [K,N]) -> Wt[n][k] f16
// ---------------------------------------------------------------------------
__global__ void k_transpose_w(const float* __restrict__ W,
                              _Float16* __restrict__ Wt, int Kdim, int Ndim) {
    int i = blockIdx.x * 256 + threadIdx.x;
    if (i < Kdim * Ndim) {
        int n = i / Kdim;
        int k = i % Kdim;
        Wt[i] = (_Float16)W[(size_t)k * Ndim + n];
    }
}

// ---------------------------------------------------------------------------
// Kernel 3: WMMA GEMM  C[M,N] = A[M,K](f16) * Bt[N,K]^T(f16) + bias (f32 out)
// block = 256 threads (8 waves). Block tile = 128 rows x 64 cols.
// wave w: rows row0 = by*128 + w*16, cols col0 = bx*64 .. +63 (4 accumulators)
// ---------------------------------------------------------------------------
__global__ void __launch_bounds__(256)
k_gemm_wmma(const _Float16* __restrict__ A, const _Float16* __restrict__ Bt,
            const float* __restrict__ bias, float* __restrict__ C,
            int Ndim, int Kdim) {
    const int wave  = threadIdx.x >> 5;
    const int lane  = threadIdx.x & 31;
    const int nl    = lane & 15;
    const int khalf = lane >> 4;
    const int col0  = blockIdx.x * 64;
    const int row0  = blockIdx.y * 128 + wave * 16;

    const _Float16* arow = A  + (size_t)(row0 + nl) * Kdim + khalf * 8;
    const _Float16* b0   = Bt + (size_t)(col0 +  0 + nl) * Kdim + khalf * 8;
    const _Float16* b1   = Bt + (size_t)(col0 + 16 + nl) * Kdim + khalf * 8;
    const _Float16* b2   = Bt + (size_t)(col0 + 32 + nl) * Kdim + khalf * 8;
    const _Float16* b3   = Bt + (size_t)(col0 + 48 + nl) * Kdim + khalf * 8;

    v8f acc0 = {}, acc1 = {}, acc2 = {}, acc3 = {};

    for (int k0 = 0; k0 < Kdim; k0 += 32) {
        __builtin_prefetch(arow + k0 + 128, 0, 0);  // global_prefetch_b8
        v16h a  = make_frag(arow + k0);
        v16h f0 = make_frag(b0 + k0);
        v16h f1 = make_frag(b1 + k0);
        v16h f2 = make_frag(b2 + k0);
        v16h f3 = make_frag(b3 + k0);
        acc0 = __builtin_amdgcn_wmma_f32_16x16x32_f16(false, a, false, f0,
                                                      (short)0, acc0, false, false);
        acc1 = __builtin_amdgcn_wmma_f32_16x16x32_f16(false, a, false, f1,
                                                      (short)0, acc1, false, false);
        acc2 = __builtin_amdgcn_wmma_f32_16x16x32_f16(false, a, false, f2,
                                                      (short)0, acc2, false, false);
        acc3 = __builtin_amdgcn_wmma_f32_16x16x32_f16(false, a, false, f3,
                                                      (short)0, acc3, false, false);
    }

    // C/D layout: lanes 0-15 -> m = r, lanes 16-31 -> m = r+8; n = col0+j*16+nl
    const int mo = khalf * 8;
    const float bz0 = bias[col0 +  0 + nl];
    const float bz1 = bias[col0 + 16 + nl];
    const float bz2 = bias[col0 + 32 + nl];
    const float bz3 = bias[col0 + 48 + nl];
#pragma unroll
    for (int r = 0; r < 8; ++r) {
        float* crow = C + (size_t)(row0 + mo + r) * Ndim + col0;
        crow[ 0 + nl] = acc0[r] + bz0;
        crow[16 + nl] = acc1[r] + bz1;
        crow[32 + nl] = acc2[r] + bz2;
        crow[48 + nl] = acc3[r] + bz3;
    }
}

// ---------------------------------------------------------------------------
// Kernel 4: V (f32 [B,T,E]) -> Vt (f16 [B,H,HD,T]) for K-contiguous B-frags
// ---------------------------------------------------------------------------
__global__ void k_transpose_v(const float* __restrict__ V,
                              _Float16* __restrict__ Vt) {
    int i = blockIdx.x * 256 + threadIdx.x;
    const int total = Bc * Hc * HDc * Tc;
    if (i < total) {
        int s  = i % Tc;
        int d  = (i / Tc) % HDc;
        int bh = i / (Tc * HDc);
        int b  = bh / Hc, h = bh % Hc;
        Vt[i] = (_Float16)V[((size_t)(b * Tc + s)) * Ec + h * HDc + d];
    }
}

// ---------------------------------------------------------------------------
// Kernel 5: fused additive score + softmax + P@V (WMMA)
// grid = (B*H, T/16), block = 128 threads (4 waves). Output f16 [B*T, E].
// ---------------------------------------------------------------------------
__global__ void __launch_bounds__(128)
k_additive_attn(const float* __restrict__ Q, const float* __restrict__ Kf,
                const _Float16* __restrict__ Vt,
                const float* __restrict__ w_score,
                const float* __restrict__ b_score,
                _Float16* __restrict__ AttnH) {
    const int bh = blockIdx.x;
    const int b  = bh / Hc, h = bh % Hc;
    const int t0 = blockIdx.y * 16;
    const int tid = threadIdx.x;

    __shared__ float    q_s[16][HDc];      //  4 KB
    __shared__ float    sc[16][Sc];        // 24 KB
    __shared__ _Float16 p_s[16][SPAD];     // ~12.25 KB (padded rows)
    __shared__ float    red[16][8];
    __shared__ float    rowstat[16];
    __shared__ float    wsc[HDc];

    if (tid < HDc) wsc[tid] = w_score[tid];
    for (int i = tid; i < 16 * HDc; i += 128) {
        int r = i / HDc, d = i % HDc;
        q_s[r][d] = Q[((size_t)(b * Tc + t0 + r)) * Ec + h * HDc + d];
    }
    __syncthreads();

    // ---- additive scores: sc[t][s] = sum_d w[d]*tanh(q[t,d]+k[s,d]) + b ----
    const float bsc = b_score[0];
    for (int p = tid; p < 16 * Sc; p += 128) {
        int t = p / Sc, s = p % Sc;
        const float* krow = Kf + ((size_t)(b * Tc + s)) * Ec + h * HDc;
        float acc = 0.f;
#pragma unroll 4
        for (int d = 0; d < HDc; d += 4) {
            float4 kv = *reinterpret_cast<const float4*>(krow + d);
            float4 qv = *reinterpret_cast<const float4*>(&q_s[t][d]);
            float4 wv = *reinterpret_cast<const float4*>(&wsc[d]);
            acc += wv.x * fast_tanh(qv.x + kv.x);
            acc += wv.y * fast_tanh(qv.y + kv.y);
            acc += wv.z * fast_tanh(qv.z + kv.z);
            acc += wv.w * fast_tanh(qv.w + kv.w);
        }
        sc[t][s] = acc + bsc;
    }
    __syncthreads();

    // ---- softmax over s, 8 threads per row ----
    const int r   = tid & 15;
    const int sub = tid >> 4;      // 0..7
    {
        float mx = -3.4e38f;
        for (int s = sub; s < Sc; s += 8) mx = fmaxf(mx, sc[r][s]);
        red[r][sub] = mx;
    }
    __syncthreads();
    if (sub == 0) {
        float mx = red[r][0];
#pragma unroll
        for (int i = 1; i < 8; ++i) mx = fmaxf(mx, red[r][i]);
        rowstat[r] = mx;
    }
    __syncthreads();
    {
        float mx = rowstat[r];
        float sum = 0.f;
        for (int s = sub; s < Sc; s += 8) {
            float e = __expf(sc[r][s] - mx);
            sc[r][s] = e;
            sum += e;
        }
        red[r][sub] = sum;
    }
    __syncthreads();
    if (sub == 0) {
        float sum = 0.f;
#pragma unroll
        for (int i = 0; i < 8; ++i) sum += red[r][i];
        rowstat[r] = 1.f / sum;
    }
    __syncthreads();
    {
        float rinv = rowstat[r];
        for (int s = sub; s < Sc; s += 8)
            p_s[r][s] = (_Float16)(sc[r][s] * rinv);
    }
    __syncthreads();

    // ---- P[16,S] @ V[S,HD] via WMMA; wave w owns d-columns w*16..w*16+15 ----
    const int wave  = tid >> 5;
    const int lane  = tid & 31;
    const int nl    = lane & 15;
    const int khalf = lane >> 4;

    v8f acc = {};
    const _Float16* prow  = &p_s[nl][khalf * 8];                     // LDS
    const _Float16* vbase = Vt + ((size_t)bh * HDc + wave * 16 + nl) * Tc
                               + khalf * 8;                           // global
    for (int k0 = 0; k0 < Sc; k0 += 32) {
        __builtin_prefetch(vbase + k0 + 64, 0, 0);
        v16h a  = make_frag(prow + k0);      // ds_read_b128 x2, conflict-free
        v16h bb = make_frag(vbase + k0);     // global_load_b128 x2
        acc = __builtin_amdgcn_wmma_f32_16x16x32_f16(false, a, false, bb,
                                                     (short)0, acc, false, false);
    }

    const int mo = khalf * 8;
#pragma unroll
    for (int rr = 0; rr < 8; ++rr) {
        AttnH[((size_t)(b * Tc + t0 + mo + rr)) * Ec + h * HDc + wave * 16 + nl]
            = (_Float16)acc[rr];
    }
}

// ---------------------------------------------------------------------------
// Host launcher
// ---------------------------------------------------------------------------
extern "C" void kernel_launch(void* const* d_in, const int* in_sizes, int n_in,
                              void* d_out, int out_size, void* d_ws, size_t ws_size,
                              hipStream_t stream) {
    const float* X   = (const float*)d_in[0];
    const float* Wq  = (const float*)d_in[1];
    const float* bq  = (const float*)d_in[2];
    const float* Wk  = (const float*)d_in[3];
    const float* bk  = (const float*)d_in[4];
    const float* Wv  = (const float*)d_in[5];
    const float* bv  = (const float*)d_in[6];
    const float* Wo  = (const float*)d_in[7];
    const float* bo  = (const float*)d_in[8];
    const float* wsc = (const float*)d_in[9];
    const float* bsc = (const float*)d_in[10];

    char* ws = (char*)d_ws;
    size_t off = 0;
    auto carve = [&](size_t bytes) {
        void* p = ws + off;
        off += (bytes + 255) & ~(size_t)255;
        return p;
    };

    _Float16* Xh    = (_Float16*)carve((size_t)Mc * Ec * 2);
    _Float16* Wqt   = (_Float16*)carve((size_t)Ec * Ec * 2);
    _Float16* Wkt   = (_Float16*)carve((size_t)Ec * Ec * 2);
    _Float16* Wvt   = (_Float16*)carve((size_t)Ec * Ec * 2);
    _Float16* Wot   = (_Float16*)carve((size_t)Ec * Ec * 2);
    float*    Qf    = (float*)   carve((size_t)Mc * Ec * 4);
    float*    Kff   = (float*)   carve((size_t)Mc * Ec * 4);
    float*    Vf    = (float*)   carve((size_t)Mc * Ec * 4);
    _Float16* Vth   = (_Float16*)carve((size_t)Bc * Hc * HDc * Tc * 2);
    _Float16* AttnH = (_Float16*)carve((size_t)Mc * Ec * 2);

    // 1) X -> f16
    {
        int n = Mc * Ec;
        k_convert_f16<<<(n + 255) / 256, 256, 0, stream>>>(X, Xh, n);
    }
    // 2) weights -> transposed f16
    {
        int n = Ec * Ec;
        dim3 g((n + 255) / 256);
        k_transpose_w<<<g, 256, 0, stream>>>(Wq, Wqt, Ec, Ec);
        k_transpose_w<<<g, 256, 0, stream>>>(Wk, Wkt, Ec, Ec);
        k_transpose_w<<<g, 256, 0, stream>>>(Wv, Wvt, Ec, Ec);
        k_transpose_w<<<g, 256, 0, stream>>>(Wo, Wot, Ec, Ec);
    }
    // 3) Q/K/V projections (WMMA GEMMs)
    {
        dim3 grid(Ec / 64, Mc / 128);
        k_gemm_wmma<<<grid, 256, 0, stream>>>(Xh, Wqt, bq, Qf,  Ec, Ec);
        k_gemm_wmma<<<grid, 256, 0, stream>>>(Xh, Wkt, bk, Kff, Ec, Ec);
        k_gemm_wmma<<<grid, 256, 0, stream>>>(Xh, Wvt, bv, Vf,  Ec, Ec);
    }
    // 4) V -> [B,H,HD,T] f16
    {
        int n = Bc * Hc * HDc * Tc;
        k_transpose_v<<<(n + 255) / 256, 256, 0, stream>>>(Vf, Vth);
    }
    // 5) fused additive score + softmax + P@V
    {
        dim3 grid(Bc * Hc, Tc / 16);
        k_additive_attn<<<grid, 128, 0, stream>>>(Qf, Kff, Vth, wsc, bsc, AttnH);
    }
    // 6) output projection -> d_out (f32)
    {
        dim3 grid(Ec / 64, Mc / 128);
        k_gemm_wmma<<<grid, 256, 0, stream>>>(AttnH, Wot, bo, (float*)d_out, Ec, Ec);
    }
}
